// Mamba3Layer_88536455840140
// MI455X (gfx1250) — matrix operand order repaired and checked by reading, not compiled, not run on hardware
//
#include <hip/hip_runtime.h>
#include <hip/hip_bf16.h>

// ---------------- problem constants ----------------
#define DMODEL   2048
#define DINNER   4096
#define NHEADS_  64
#define DSTATE   64
#define HEADDIM_ 64
#define BATCH_   2
#define SEQ_     4096
#define MTOT     (BATCH_ * SEQ_)          // 8192 rows
#define NEFF     (DINNER + NHEADS_ + DSTATE + DSTATE + NHEADS_)  // 4352
// column offsets inside proj_eff
#define COL_Z    0
#define COL_XS   DINNER                    // 4096
#define COL_B    (DINNER + NHEADS_)        // 4160
#define COL_C    (DINNER + NHEADS_ + DSTATE)       // 4224
#define COL_DT   (DINNER + NHEADS_ + 2*DSTATE)     // 4288

typedef __attribute__((ext_vector_type(16))) __bf16 v16bf;
typedef __attribute__((ext_vector_type(8)))  float  v8f;
typedef int v4i_ __attribute__((vector_size(16)));

union FragBf { v16bf v; uint4 q[2]; };

#if __has_builtin(__builtin_amdgcn_global_load_async_to_lds_b128)
#define HAVE_ASYNC_LDS 1
#else
#define HAVE_ASYNC_LDS 0
#endif

// 16-byte global -> LDS copy; async (GLOBAL_LOAD_ASYNC_TO_LDS_B128) when the
// builtin exists, otherwise load+ds_store.
static __device__ __forceinline__ void cp16(const unsigned short* g, unsigned short* l) {
#if HAVE_ASYNC_LDS
    __builtin_amdgcn_global_load_async_to_lds_b128(
        (__attribute__((address_space(1))) v4i_*)g,
        (__attribute__((address_space(3))) v4i_*)l,
        0, 0);
#else
    *(uint4*)l = *(const uint4*)g;
#endif
}

static __device__ __forceinline__ void wait_async_lds() {
#if HAVE_ASYNC_LDS
#if __has_builtin(__builtin_amdgcn_s_wait_asynccnt)
    __builtin_amdgcn_s_wait_asynccnt(0);
#else
    asm volatile("s_wait_asynccnt 0" ::: "memory");
#endif
#endif
}

// fp32 -> bf16 (round to nearest even)
static __device__ __forceinline__ unsigned short f2bf(float f) {
    unsigned int u = __float_as_uint(f);
    u += 0x7FFFu + ((u >> 16) & 1u);
    return (unsigned short)(u >> 16);
}

// ---------------- conversion kernels ----------------
__global__ __launch_bounds__(256) void k_f32_to_bf16(const float* __restrict__ in,
                                                     unsigned short* __restrict__ out,
                                                     int n) {
    int i = blockIdx.x * 256 + threadIdx.x;
    if (i < n) out[i] = f2bf(in[i]);
}

// Build W_eff (NEFF x DMODEL, bf16) from W_in (8384 x DMODEL, fp32),
// folding the per-head mean over HEADDIM of the x_ssm block into 64 "xs" rows.
__global__ __launch_bounds__(256) void k_build_weff(const float* __restrict__ W_in,
                                                    unsigned short* __restrict__ W_eff) {
    int i = blockIdx.x * 256 + threadIdx.x;
    if (i >= NEFF * DMODEL) return;
    int r = i / DMODEL;
    int c = i - r * DMODEL;
    float v;
    if (r < DINNER) {                       // z rows: passthrough
        v = W_in[(size_t)r * DMODEL + c];
    } else if (r < DINNER + NHEADS_) {      // xs rows: mean of head block
        int h = r - DINNER;
        const float* base = W_in + ((size_t)(DINNER + h * HEADDIM_)) * DMODEL + c;
        float s = 0.f;
        #pragma unroll 8
        for (int j = 0; j < HEADDIM_; ++j) s += base[(size_t)j * DMODEL];
        v = s * (1.0f / (float)HEADDIM_);
    } else {                                // B, C, dt rows: shifted passthrough
        v = W_in[(size_t)(r + (DINNER - NHEADS_)) * DMODEL + c]; // r+4032 -> 8192..8383
    }
    W_eff[i] = f2bf(v);
}

// ---------------- WMMA bf16 GEMM:  C(MxN,f32) = A(MxK,bf16) * B(NxK,bf16)^T --------
// Block: 256 threads = 8 waves (2x4); block tile 128x256; wave tile 64x64
// (4x4 WMMA 16x16x32). LDS double-buffered staging, KC=32, row pitch 40
// ushorts (80B) to break bank-conflict periodicity.
#define KC      32
#define PITCH   40                      // ushorts per staged row (32 data + 8 pad)

__global__ __launch_bounds__(256)
void k_gemm_bf16_wmma(const unsigned short* __restrict__ A,
                      const unsigned short* __restrict__ Bm,
                      float* __restrict__ C,
                      int M, int N, int K) {
    __shared__ alignas(16) unsigned short sA[2][128 * PITCH]; // 2 x 10240 B
    __shared__ alignas(16) unsigned short sB[2][256 * PITCH]; // 2 x 20480 B

    const int tid  = threadIdx.x;
    const int lane = tid & 31;
    const int wave = tid >> 5;
    const int wm   = wave >> 2;            // 0..1
    const int wn   = wave & 3;             // 0..3
    const int mBlock = blockIdx.x * 128;
    const int nBlock = blockIdx.y * 256;
    const int half = lane >> 4;            // 0/1
    const int sub  = lane & 15;            // row/col within a 16-tile

    // cooperative staging of one KC-slice into buffer `buf`
    auto stage = [&](int buf, int kpos) {
        #pragma unroll
        for (int c = 0; c < 512; c += 256) {            // A: 512 x 16B chunks
            int ch  = c + tid;
            int row = ch >> 2;
            int kp  = (ch & 3) * 8;
            cp16(A + (size_t)(mBlock + row) * (size_t)K + kpos + kp,
                 &sA[buf][row * PITCH + kp]);
        }
        #pragma unroll
        for (int c = 0; c < 1024; c += 256) {           // B: 1024 x 16B chunks
            int ch  = c + tid;
            int row = ch >> 2;
            int kp  = (ch & 3) * 8;
            cp16(Bm + (size_t)(nBlock + row) * (size_t)K + kpos + kp,
                 &sB[buf][row * PITCH + kp]);
        }
    };

    v8f acc[4][4];
    #pragma unroll
    for (int i = 0; i < 4; ++i)
        #pragma unroll
        for (int j = 0; j < 4; ++j) acc[i][j] = (v8f){0.f,0.f,0.f,0.f,0.f,0.f,0.f,0.f};

    stage(0, 0);
    wait_async_lds();
    __syncthreads();

    const int nk = K / KC;
    for (int kk = 0; kk < nk; ++kk) {
        const int cur = kk & 1;
        if (kk + 1 < nk) stage(cur ^ 1, (kk + 1) * KC);

        // fragment reads from LDS (ds_load_b128, immediate offsets)
        // A lane layout (16x32 bf16): K = 8*half+0..7 then 16+8*half+0..7
        // B lane layout (32x16 bf16): K = 8*half+0..7 then 16+8*half+0..7 of column `sub`
        const unsigned short* aBase = &sA[cur][(wm * 64 + sub) * PITCH + 8 * half];
        const unsigned short* bBase = &sB[cur][(wn * 64 + sub) * PITCH + 8 * half];
        FragBf a[4], b[4];
        #pragma unroll
        for (int i = 0; i < 4; ++i) {
            a[i].q[0] = *(const uint4*)(aBase + i * 16 * PITCH);
            a[i].q[1] = *(const uint4*)(aBase + i * 16 * PITCH + 16);
        }
        #pragma unroll
        for (int j = 0; j < 4; ++j) {
            b[j].q[0] = *(const uint4*)(bBase + j * 16 * PITCH);
            b[j].q[1] = *(const uint4*)(bBase + j * 16 * PITCH + 16);
        }
        #pragma unroll
        for (int i = 0; i < 4; ++i)
            #pragma unroll
            for (int j = 0; j < 4; ++j)
                acc[i][j] = __builtin_amdgcn_wmma_f32_16x16x32_bf16(
                    /*neg_a=*/false, a[i].v, /*neg_b=*/false, b[j].v,
                    /*c_mod=*/(short)0, acc[i][j],
                    /*reuse_a=*/false, /*reuse_b=*/false);

        wait_async_lds();
        __syncthreads();
    }

    // C/D layout: VGPR v -> row (v + 8*half), col = sub  (ISA 7.12.2)
    #pragma unroll
    for (int i = 0; i < 4; ++i) {
        const size_t row0 = (size_t)(mBlock + wm * 64 + i * 16 + 8 * half);
        #pragma unroll
        for (int j = 0; j < 4; ++j) {
            float* cp = C + row0 * (size_t)N + (size_t)(nBlock + wn * 64 + j * 16 + sub);
            #pragma unroll
            for (int v = 0; v < 8; ++v) cp[(size_t)v * N] = acc[i][j][v];
        }
    }
}

// ---------------- SSM scan: one wave32 per (b,h), 2 states per lane --------------
__global__ __launch_bounds__(128)
void k_scan(const float* __restrict__ proj,   // (MTOT x NEFF)
            const float* __restrict__ A_log,
            const float* __restrict__ Dp,
            const float* __restrict__ dt_bias,
            float* __restrict__ y_head) {     // (MTOT x NHEADS_)
    const int lane = threadIdx.x & 31;
    const int wg   = blockIdx.x * 4 + (threadIdx.x >> 5);  // 0..127
    const int b    = wg >> 6;           // /NHEADS_
    const int h    = wg & (NHEADS_ - 1);

    const float Ah  = -__expf(A_log[h]);
    const float Dh  = Dp[h];
    const float dtb = dt_bias[h];

    float h0 = 0.f, h1 = 0.f;           // states lane and lane+32

    const float* row = proj + (size_t)b * SEQ_ * NEFF;
    for (int t = 0; t < SEQ_; ++t, row += NEFF) {
        __builtin_prefetch(row + NEFF + COL_XS, 0, 1);   // next timestep row

        const float xs    = row[COL_XS + h];
        const float dtraw = row[COL_DT + h] + dtb;
        const float Bv0   = row[COL_B + lane];
        const float Bv1   = row[COL_B + lane + 32];
        const float Cv0   = row[COL_C + lane];
        const float Cv1   = row[COL_C + lane + 32];

        const float dt    = (dtraw > 20.f) ? dtraw : log1pf(__expf(dtraw));
        const float Adt   = Ah * dt;
        // h_new = h*(ALPHA*(1+dt*A) + (1-ALPHA)*exp(dt*A)) + dt*B*xs  (ALPHA=0.5)
        const float coef  = 0.5f * (1.f + Adt) + 0.5f * __expf(Adt);
        const float dx    = dt * xs;
        h0 = coef * h0 + dx * Bv0;
        h1 = coef * h1 + dx * Bv1;

        float part = Cv0 * h0 + Cv1 * h1;
        #pragma unroll
        for (int off = 16; off > 0; off >>= 1)
            part += __shfl_xor(part, off, 32);

        if (lane == 0)
            y_head[((size_t)b * SEQ_ + t) * NHEADS_ + h] = part + Dh * xs;
    }
}

// ---------------- gating: gated[m,i] = bf16( y_head[m, i/64] * silu(z[m,i]) ) -----
__global__ __launch_bounds__(256)
void k_gate(const float* __restrict__ proj,
            const float* __restrict__ y_head,
            unsigned short* __restrict__ gated) {
    int i = blockIdx.x * 256 + threadIdx.x;
    if (i >= MTOT * DINNER) return;
    int m = i / DINNER;
    int c = i - m * DINNER;
    float z = proj[(size_t)m * NEFF + c];
    float y = y_head[(size_t)m * NHEADS_ + (c >> 6)];   // HEADDIM_=64
    float s = z / (1.f + __expf(-z));                   // silu
    gated[i] = f2bf(y * s);
}

// ---------------- launcher ----------------
extern "C" void kernel_launch(void* const* d_in, const int* in_sizes, int n_in,
                              void* d_out, int out_size, void* d_ws, size_t ws_size,
                              hipStream_t stream) {
    const float* x       = (const float*)d_in[0];   // (B,L,DMODEL)
    const float* W_in    = (const float*)d_in[1];   // (8384, DMODEL)
    const float* A_log   = (const float*)d_in[2];   // (NHEADS)
    const float* Dp      = (const float*)d_in[3];   // (NHEADS)
    const float* dt_bias = (const float*)d_in[4];   // (NHEADS)
    const float* W_out   = (const float*)d_in[5];   // (DMODEL, DINNER)
    float* out           = (float*)d_out;           // (B,L,DMODEL)

    char* ws = (char*)d_ws;
    unsigned short* xb    = (unsigned short*)(ws + 0);                 // 33.6 MB
    unsigned short* Weff  = (unsigned short*)(ws + 33554432);          // 17.8 MB
    unsigned short* Wob   = (unsigned short*)(ws + 51380224);          // 16.8 MB
    float*          proj  = (float*)        (ws + 68157440);           // 142.6 MB
    float*          yhead = (float*)        (ws + 210763776);          // 2.1 MB
    unsigned short* gated = (unsigned short*)(ws + 212860928);         // 67.1 MB

    // 1) convert x -> bf16
    {
        int n = MTOT * DMODEL;
        k_f32_to_bf16<<<(n + 255) / 256, 256, 0, stream>>>(x, xb, n);
    }
    // 2) build effective projection weights (z | mean-folded xs | B | C | dt)
    {
        int n = NEFF * DMODEL;
        k_build_weff<<<(n + 255) / 256, 256, 0, stream>>>(W_in, Weff);
    }
    // 3) convert W_out -> bf16
    {
        int n = DMODEL * DINNER;
        k_f32_to_bf16<<<(n + 255) / 256, 256, 0, stream>>>(W_out, Wob, n);
    }
    // 4) GEMM1: proj(8192 x 4352) = xb(8192x2048) * Weff(4352x2048)^T
    {
        dim3 grid(MTOT / 128, NEFF / 256);   // 64 x 17
        k_gemm_bf16_wmma<<<grid, 256, 0, stream>>>(xb, Weff, proj, MTOT, NEFF, DMODEL);
    }
    // 5) sequential SSM scan -> y_head
    {
        k_scan<<<32, 128, 0, stream>>>(proj, A_log, Dp, dt_bias, yhead);
    }
    // 6) gate with silu(z), convert to bf16
    {
        int n = MTOT * DINNER;
        k_gate<<<(n + 255) / 256, 256, 0, stream>>>(proj, yhead, gated);
    }
    // 7) GEMM2: out(8192 x 2048) = gated(8192x4096) * Wob(2048x4096)^T
    {
        dim3 grid(MTOT / 128, DMODEL / 256); // 64 x 8
        k_gemm_bf16_wmma<<<grid, 256, 0, stream>>>(gated, Wob, out, MTOT, DMODEL, DINNER);
    }
}
